// RNN_18511309046075
// MI455X (gfx1250) — matrix operand order repaired and checked by reading, not compiled, not run on hardware
//
#include <hip/hip_runtime.h>
#include <hip/hip_bf16.h>

// ---------------------------------------------------------------------------
// CDNA5 (gfx1250) implementation: wave32 + V_WMMA_F32_16X16X32_F16 tiles.
// Weights pre-packed (once per call) into zero-padded f16 matrices so GEMM
// inner loops are pure {ds_load_b128 A-frag, global_load_b128 f16 B-frag, wmma}.
// pred2 stages its f16 activation tile with global_load_async_to_lds_b128
// (ASYNCcnt path).  Activations use native v_exp_f32 / v_rcp_f32.
// ---------------------------------------------------------------------------

#define B_   8192
#define T_   16
#define IN_  120
#define CLS_ 1000
#define HID_ 60

typedef __attribute__((ext_vector_type(16))) _Float16 v16h;
typedef __attribute__((ext_vector_type(8)))  _Float16 v8h;
typedef __attribute__((ext_vector_type(8)))  float    v8f;

__device__ __forceinline__ v8f wmma16(v16h a, v16h b, v8f c) {
    return __builtin_amdgcn_wmma_f32_16x16x32_f16(
        false, a, false, b, (short)0, c, false, false);
}

// A fragment: 16x32 f16 tile, row-major [16][stride] in LDS.
// lanes 0-15: M=0..15, K = 0..15 ; lanes 16-31: K = 16..31.
__device__ __forceinline__ v16h afrag(const _Float16* p, int stride) {
    int l = threadIdx.x & 31;
    int r = l & 15, kh = (l >> 4) * 16;
    const _Float16* q = p + r * stride + kh;
    v8h c0 = *(const v8h*)q;
    v8h c1 = *(const v8h*)(q + 8);
    v16h f;
#pragma unroll
    for (int i = 0; i < 8; i++) { f[i] = c0[i]; f[8 + i] = c1[i]; }
    return f;
}

// B fragment from a pre-packed f16 weight Wp[outP][KP] (zero padded -> no
// guards).  lane l -> column col0+(l&15); 16 contiguous K at k0+(l>>4)*16.
__device__ __forceinline__ v16h bfrag_p(const _Float16* __restrict__ Wp,
                                        int KP, int col0, int k0) {
    int l = threadIdx.x & 31;
    const _Float16* p = Wp + (size_t)(col0 + (l & 15)) * KP + k0 + ((l >> 4) << 4);
    v8h c0 = *(const v8h*)p;
    v8h c1 = *(const v8h*)(p + 8);
    v16h f;
#pragma unroll
    for (int i = 0; i < 8; i++) { f[i] = c0[i]; f[8 + i] = c1[i]; }
    return f;
}

// Branchless fast activations: native v_exp_f32 (via __expf) + v_rcp_f32.
__device__ __forceinline__ float rcpf(float x) { return __builtin_amdgcn_rcpf(x); }
__device__ __forceinline__ float sigf(float x) { return rcpf(1.f + __expf(-x)); }
__device__ __forceinline__ float tanh_f(float x) {
    return 1.f - 2.f * rcpf(__expf(2.f * x) + 1.f);
}

// Async global -> LDS copy of 16 bytes (gfx1250, tracked by ASYNCcnt).
// lds_off: byte offset within the wave's LDS allocation (low 32 bits of the
// flat shared address == LDS offset per the aperture mapping).
__device__ __forceinline__ void async_g2l_b128(unsigned lds_off, const void* g) {
    asm volatile("global_load_async_to_lds_b128 %0, %1, off"
                 :: "v"(lds_off), "v"(g)
                 : "memory");
}
__device__ __forceinline__ void wait_asynccnt0() {
    asm volatile("s_wait_asynccnt 0x0" ::: "memory");
}

// ---------------------------------------------------------------------------
// Weight pre-packing (runs once per call; weights are loop-invariant).
// ---------------------------------------------------------------------------
__global__ __launch_bounds__(256)
void pack_w_kernel(const float* __restrict__ W, int out, int kin,
                   _Float16* __restrict__ Wp, int outP, int KP)
{
    int idx = blockIdx.x * 256 + threadIdx.x;
    if (idx >= outP * KP) return;
    int o = idx / KP, k = idx % KP;
    float v = (o < out && k < kin) ? W[(size_t)o * kin + k] : 0.f;
    Wp[idx] = (_Float16)v;
}

// Concatenated LSTM weight: k<kin -> Wih[out][kin], kin<=k<kin+khid -> Whh
__global__ __launch_bounds__(256)
void pack_w2_kernel(const float* __restrict__ Wih, int kin,
                    const float* __restrict__ Whh, int khid,
                    _Float16* __restrict__ Wp, int out, int KP)
{
    int idx = blockIdx.x * 256 + threadIdx.x;
    if (idx >= out * KP) return;
    int o = idx / KP, k = idx % KP;
    float v = 0.f;
    if (k < kin)             v = Wih[(size_t)o * kin + k];
    else if (k < kin + khid) v = Whh[(size_t)o * khid + (k - kin)];
    Wp[idx] = (_Float16)v;
}

// ---------------------------------------------------------------------------
// _feat: y = W2 @ tanh(bn2(W1 @ bn1(x) + b1)) + b2   (120 -> 240 -> 120)
// W1p: 240x128 f16, W2p: 128x256 f16.
// Row remap: row r -> offset base + ((r/grp)*gstride + r%grp)*120
// ---------------------------------------------------------------------------
__global__ __launch_bounds__(256)
void feat_kernel(const float* __restrict__ X, long long xbase, int xgrp, int xgs,
                 int nrows,
                 const float* bn1b, const float* bn1g, const float* bn1m, const float* bn1v,
                 const _Float16* __restrict__ W1p, const float* b1,
                 const float* bn2b, const float* bn2g, const float* bn2m, const float* bn2v,
                 const _Float16* __restrict__ W2p, const float* b2,
                 float* __restrict__ Y, long long ybase, int ygrp, int ygs)
{
    __shared__ float s1[128], t1[128], s2[240], t2[240];
    __shared__ __align__(16) _Float16 Xt[64 * 128];
    __shared__ __align__(16) _Float16 Ht[64 * 256];
    int tid = threadIdx.x;
    int row0 = blockIdx.x * 64;

    if (tid < 128) {
        if (tid < 120) {
            float sv = bn1g[tid] * rsqrtf(bn1v[tid] + 1e-5f);
            s1[tid] = sv; t1[tid] = bn1b[tid] - bn1m[tid] * sv;
        } else { s1[tid] = 0.f; t1[tid] = 0.f; }
    }
    if (tid < 240) {
        float sv = bn2g[tid] * rsqrtf(bn2v[tid] + 1e-5f);
        s2[tid] = sv; t2[tid] = bn2b[tid] - bn2m[tid] * sv;
    }
    __syncthreads();

    for (int idx = tid; idx < 64 * 128; idx += 256) {
        int r = idx >> 7, k = idx & 127;
        int row = row0 + r;
        float val = 0.f;
        if (k < 120 && row < nrows) {
            long long xo = xbase + ((long long)(row / xgrp) * xgs + (row % xgrp)) * 120;
            val = X[xo + k] * s1[k] + t1[k];
        }
        Xt[idx] = (_Float16)val;
    }
    for (int idx = tid; idx < 64 * 16; idx += 256)          // zero pad cols 240..255
        Ht[(idx >> 4) * 256 + 240 + (idx & 15)] = (_Float16)0.f;
    __syncthreads();

    int w = tid >> 5, lane = tid & 31;
    int n = lane & 15, mhi = (lane >> 4) * 8;

    // GEMM1: (64x128) x (128x240).  Wave owns col tiles j=w, w+8; B frags
    // preloaded once and reused across the 4 row tiles.
    for (int j = w; j < 15; j += 8) {
        v16h bf[4];
#pragma unroll
        for (int kt = 0; kt < 4; kt++) bf[kt] = bfrag_p(W1p, 128, j * 16, kt * 32);
        int col = j * 16 + n;
        float b1c = b1[col], sc = s2[col], tc = t2[col];
#pragma unroll
        for (int rt = 0; rt < 4; rt++) {
            v8f acc = {};
#pragma unroll
            for (int kt = 0; kt < 4; kt++)
                acc = wmma16(afrag(&Xt[rt * 16 * 128 + kt * 32], 128), bf[kt], acc);
#pragma unroll
            for (int v = 0; v < 8; v++) {
                float z = tanh_f((acc[v] + b1c) * sc + tc);
                Ht[(rt * 16 + v + mhi) * 256 + col] = (_Float16)z;
            }
        }
    }
    __syncthreads();

    // GEMM2: (64x256) x (256x120).  8 col tiles -> one per wave.
    {
        int j = w;
        v16h bf[8];
#pragma unroll
        for (int kt = 0; kt < 8; kt++) bf[kt] = bfrag_p(W2p, 256, j * 16, kt * 32);
        int col = j * 16 + n;
        float b2c = (col < 120) ? b2[col] : 0.f;
#pragma unroll
        for (int rt = 0; rt < 4; rt++) {
            v8f acc = {};
#pragma unroll
            for (int kt = 0; kt < 8; kt++)
                acc = wmma16(afrag(&Ht[rt * 16 * 256 + kt * 32], 256), bf[kt], acc);
            if (col < 120) {
#pragma unroll
                for (int v = 0; v < 8; v++) {
                    int row = row0 + rt * 16 + v + mhi;
                    if (row < nrows) {
                        long long yo = ybase + ((long long)(row / ygrp) * ygs + (row % ygrp)) * 120;
                        Y[yo + col] = acc[v] + b2c;
                    }
                }
            }
        }
    }
}

// ---------------------------------------------------------------------------
// LSTM cell: gates = [x0|x1|x2|h] @ Wp.T + b_ih + b_hh  (Wp packed 240xKP)
// Templated on KP so the A-tile fill uses constant div/mod and kt unrolls.
// ---------------------------------------------------------------------------
template <int KP>
__global__ __launch_bounds__(256)
void lstm_kernel(const float* x0, long long x0b, int x0s, int len0,
                 const float* x1, long long x1b, int x1s, int len1,
                 const float* x2, long long x2b, int x2s, int len2,
                 float* __restrict__ h, float* __restrict__ c,
                 const _Float16* __restrict__ Wp,
                 const float* bih, const float* bhh,
                 int kin, int nrows,
                 float* Hs, long long hsb, int hss)
{
    __shared__ __align__(16) _Float16 At[32 * KP];
    __shared__ float gates[32 * 240];
    int tid = threadIdx.x, row0 = blockIdx.x * 32;
    int ktot = kin + 60;

    for (int idx = tid; idx < 32 * KP; idx += 256) {
        int r = idx / KP, k = idx % KP;
        int row = row0 + r;
        float val = 0.f;
        if (row < nrows) {
            if (k < len0)                      val = x0[x0b + (long long)row * x0s + k];
            else if (k < len0 + len1)          val = x1[x1b + (long long)row * x1s + (k - len0)];
            else if (k < len0 + len1 + len2)   val = x2[x2b + (long long)row * x2s + (k - len0 - len1)];
            else if (k < ktot)                 val = h[(long long)row * 60 + (k - kin)];
        }
        At[idx] = (_Float16)val;
    }
    __syncthreads();

    int w = tid >> 5, lane = tid & 31;
    int n = lane & 15, mhi = (lane >> 4) * 8;
    int rt = w & 1;
    for (int j = (w >> 1); j < 15; j += 4) {
        v8f acc = {};
#pragma unroll
        for (int kt = 0; kt < KP / 32; kt++) {
            v16h a = afrag(&At[rt * 16 * KP + kt * 32], KP);
            v16h b = bfrag_p(Wp, KP, j * 16, kt * 32);
            acc = wmma16(a, b, acc);
        }
        int col = j * 16 + n;
        float bsum = bih[col] + bhh[col];
#pragma unroll
        for (int v = 0; v < 8; v++)
            gates[(rt * 16 + v + mhi) * 240 + col] = acc[v] + bsum;
    }
    __syncthreads();

    for (int idx = tid; idx < 32 * 60; idx += 256) {
        int r = idx / 60, d = idx % 60;
        int row = row0 + r;
        if (row < nrows) {
            const float* g = &gates[r * 240];
            float gi = g[d], gf = g[60 + d], gg = g[120 + d], go = g[180 + d];
            float cold = c[(long long)row * 60 + d];
            float c2 = sigf(gf) * cold + sigf(gi) * tanh_f(gg);
            float h2 = sigf(go) * tanh_f(c2);
            c[(long long)row * 60 + d] = c2;
            h[(long long)row * 60 + d] = h2;
            if (Hs) Hs[hsb + (long long)row * hss + d] = h2;
        }
    }
}

// ---------------------------------------------------------------------------
// Attention: logits[b,tau] = w2 . relu(bn(W1 @ [h_b | H_{b,tau}] + b1))
//            Ci[b] = softmax(logits) @ H[b]      (4 batches per block)
// W1p packed 96x128 f16.
// ---------------------------------------------------------------------------
__global__ __launch_bounds__(256)
void attn_kernel(const float* __restrict__ H, const float* __restrict__ h,
                 const float* bnb, const float* bng, const float* bnm, const float* bnv,
                 const _Float16* __restrict__ W1p, const float* b1,
                 const float* w2vec, float* __restrict__ Ci)
{
    __shared__ __align__(16) _Float16 At[64 * 128];
    __shared__ float Hst[4 * 16 * 60];
    __shared__ float hst[4 * 60];
    __shared__ float s[96], t[96], w2[96];
    __shared__ float logits[64], wsm[64], sm[4];
    int tid = threadIdx.x;
    int b0 = blockIdx.x * 4;

    if (tid < 96) {
        if (tid < 90) {
            float sv = bng[tid] * rsqrtf(bnv[tid] + 1e-5f);
            s[tid] = sv; t[tid] = bnb[tid] - bnm[tid] * sv; w2[tid] = w2vec[tid];
        } else { s[tid] = 0.f; t[tid] = 0.f; w2[tid] = 0.f; }
    }
    if (tid < 64) logits[tid] = 0.f;
    for (int idx = tid; idx < 4 * 16 * 60; idx += 256)
        Hst[idx] = H[(long long)b0 * 960 + idx];
    for (int idx = tid; idx < 240; idx += 256)
        hst[idx] = h[(long long)b0 * 60 + idx];
    __syncthreads();

    for (int idx = tid; idx < 64 * 128; idx += 256) {
        int r = idx >> 7, k = idx & 127;
        int bb = r >> 4, tau = r & 15;
        float val = 0.f;
        if (k < 60)       val = hst[bb * 60 + k];
        else if (k < 120) val = Hst[bb * 960 + tau * 60 + (k - 60)];
        At[idx] = (_Float16)val;
    }
    __syncthreads();

    int w = tid >> 5, lane = tid & 31;
    int n = lane & 15, mhi = (lane >> 4) * 8;
    // 6 col tiles; wave owns j = w (waves 6,7 idle here), reuse B over 4 rtiles
    if (w < 6) {
        int j = w;
        v16h bf[4];
#pragma unroll
        for (int kt = 0; kt < 4; kt++) bf[kt] = bfrag_p(W1p, 128, j * 16, kt * 32);
        int col = j * 16 + n;
        float wc = 0.f, sc = 0.f, tc = 0.f, b1c = 0.f;
        if (col < 90) { wc = w2[col]; sc = s[col]; tc = t[col]; b1c = b1[col]; }
#pragma unroll
        for (int rt = 0; rt < 4; rt++) {
            v8f acc = {};
#pragma unroll
            for (int kt = 0; kt < 4; kt++)
                acc = wmma16(afrag(&At[rt * 16 * 128 + kt * 32], 128), bf[kt], acc);
            if (col < 90) {
#pragma unroll
                for (int v = 0; v < 8; v++) {
                    float z = (acc[v] + b1c) * sc + tc;
                    z = z > 0.f ? z : 0.f;
                    atomicAdd(&logits[rt * 16 + v + mhi], z * wc);  // ds_add_f32
                }
            }
        }
    }
    __syncthreads();

    if (tid < 4) {                                   // softmax per batch
        float m = -1e30f;
        for (int i = 0; i < 16; i++) m = fmaxf(m, logits[tid * 16 + i]);
        float ssum = 0.f;
        for (int i = 0; i < 16; i++) {
            float e = __expf(logits[tid * 16 + i] - m);
            wsm[tid * 16 + i] = e; ssum += e;
        }
        sm[tid] = rcpf(ssum);
    }
    __syncthreads();

    for (int idx = tid; idx < 240; idx += 256) {
        int bb = idx / 60, d = idx % 60;
        float acc2 = 0.f;
        for (int tau = 0; tau < 16; tau++)
            acc2 += wsm[bb * 16 + tau] * Hst[bb * 960 + tau * 60 + d];
        Ci[(long long)(b0 + bb) * 60 + d] = acc2 * sm[bb];
    }
}

// ---------------------------------------------------------------------------
// pred stage 1: Z = tanh(bn(W1 @ [yh|h2|tr|Ci] + b1))  (360 -> 240), W1p 240x384
// Z stored as f16 (it is consumed as the f16 A operand of pred2).
// ---------------------------------------------------------------------------
__global__ __launch_bounds__(256)
void pred1_kernel(const float* __restrict__ yh, const float* __restrict__ h,
                  const float* __restrict__ tr, long long trb, int trs,
                  const float* __restrict__ Cin,
                  const float* bnb, const float* bng, const float* bnm, const float* bnv,
                  const _Float16* __restrict__ W1p, const float* b1,
                  _Float16* __restrict__ Z, int nrows)
{
    __shared__ __align__(16) _Float16 At[32 * 384];
    __shared__ float s[240], t[240];
    int tid = threadIdx.x, row0 = blockIdx.x * 32;
    if (tid < 240) {
        float sv = bng[tid] * rsqrtf(bnv[tid] + 1e-5f);
        s[tid] = sv; t[tid] = bnb[tid] - bnm[tid] * sv;
    }
    for (int idx = tid; idx < 32 * 384; idx += 256) {
        int r = idx / 384, k = idx % 384;
        int row = row0 + r;
        float val = 0.f;
        if (row < nrows) {
            if (k < 120)      val = yh[(long long)row * 120 + k];
            else if (k < 180) val = h[(long long)row * 60 + (k - 120)];
            else if (k < 300) val = tr[trb + (long long)row * trs + (k - 180)];
            else if (k < 360) val = Cin[(long long)row * 60 + (k - 300)];
        }
        At[idx] = (_Float16)val;
    }
    __syncthreads();

    int w = tid >> 5, lane = tid & 31;
    int n = lane & 15, mhi = (lane >> 4) * 8;
    int rt = w & 1;
    for (int j = (w >> 1); j < 15; j += 4) {
        v8f acc = {};
#pragma unroll
        for (int kt = 0; kt < 12; kt++) {
            v16h a = afrag(&At[rt * 16 * 384 + kt * 32], 384);
            v16h b = bfrag_p(W1p, 384, j * 16, kt * 32);
            acc = wmma16(a, b, acc);
        }
        int col = j * 16 + n;
        float b1c = b1[col], sc = s[col], tc = t[col];
#pragma unroll
        for (int v = 0; v < 8; v++) {
            int row = row0 + rt * 16 + v + mhi;
            if (row < nrows)
                Z[(long long)row * 240 + col] = (_Float16)tanh_f((acc[v] + b1c) * sc + tc);
        }
    }
}

// ---------------------------------------------------------------------------
// pred stage 2: out[:,t,:] = W2 @ Z + b2  (240 -> 1000), W2p packed 1008x256.
// A-tile (64x240 f16) staged via global_load_async_to_lds_b128 (ASYNCcnt).
// ---------------------------------------------------------------------------
__global__ __launch_bounds__(256)
void pred2_kernel(const _Float16* __restrict__ Z, const _Float16* __restrict__ W2p,
                  const float* b2, float* __restrict__ out,
                  long long obase, int ostride, int nrows)
{
    __shared__ __align__(16) _Float16 At[64 * 256];
    int tid = threadIdx.x, row0 = blockIdx.x * 64;

    // zero-pad cols 240..255 (ds_store)
    for (int idx = tid; idx < 64 * 16; idx += 256)
        At[(idx >> 4) * 256 + 240 + (idx & 15)] = (_Float16)0.f;

    // async copy 64 rows x 240 halves in 16B chunks: 64*30 = 1920 transfers
    unsigned base_lds = (unsigned)(size_t)&At[0];   // LDS offset (low 32 bits)
    for (int idx = tid; idx < 64 * 30; idx += 256) {
        int r = idx / 30, ck = idx % 30;
        unsigned lds_off = base_lds + (unsigned)((r * 256 + ck * 8) * 2);
        const _Float16* g = Z + (size_t)(row0 + r) * 240 + ck * 8;
        async_g2l_b128(lds_off, g);
    }
    wait_asynccnt0();
    __syncthreads();

    int w = tid >> 5, lane = tid & 31;
    int n = lane & 15, mhi = (lane >> 4) * 8;
    for (int j = w; j < 63; j += 8) {               // 63 col tiles of 16
        v16h bf[8];
#pragma unroll
        for (int kt = 0; kt < 8; kt++) bf[kt] = bfrag_p(W2p, 256, j * 16, kt * 32);
        int col = j * 16 + n;
        float b2c = (col < 1000) ? b2[col] : 0.f;
#pragma unroll
        for (int rt = 0; rt < 4; rt++) {
            v8f acc = {};
#pragma unroll
            for (int kt = 0; kt < 8; kt++)
                acc = wmma16(afrag(&At[rt * 16 * 256 + kt * 32], 256), bf[kt], acc);
            if (col < 1000) {
#pragma unroll
                for (int v = 0; v < 8; v++) {
                    int row = row0 + rt * 16 + v + mhi;
                    if (row < nrows)
                        out[obase + (long long)row * ostride + col] = acc[v] + b2c;
                }
            }
        }
    }
}

// ---------------------------------------------------------------------------
// argmax over 1000 classes (first-max tie-break) + gather ref_feat[topi]
// ---------------------------------------------------------------------------
__global__ __launch_bounds__(256)
void argmax_gather_kernel(const float* __restrict__ out, long long obase, int ostride,
                          const float* __restrict__ ref, float* __restrict__ Y, int nrows)
{
    int w = threadIdx.x >> 5, lane = threadIdx.x & 31;
    int row = blockIdx.x * 8 + w;
    if (row >= nrows) return;
    const float* p = out + obase + (long long)row * ostride;
    float best = -1e30f; int bi = 0;
    for (int j = lane; j < 1000; j += 32) {
        float v = p[j];
        if (v > best) { best = v; bi = j; }
    }
#pragma unroll
    for (int off = 16; off > 0; off >>= 1) {
        float ov = __shfl_down(best, off, 32);
        int   oi = __shfl_down(bi, off, 32);
        if (ov > best || (ov == best && oi < bi)) { best = ov; bi = oi; }
    }
    bi = __shfl(bi, 0, 32);
    const float* rrow = ref + (long long)bi * (IN_ + 3);
    for (int j = lane; j < 120; j += 32)
        Y[(long long)row * 120 + j] = rrow[j];
}

// ---------------------------------------------------------------------------
// Host orchestration.  Input order assumes JAX pytree (sorted-key) flattening:
//  0 RSSI  1 DRSSI  2 refrence_rssi
//  dec:     3 W_hh 4 W_ih 5 b_hh 6 b_ih
//  drssi2h: 7..10 bn1{b,g,m,v} 11..14 bn2{b,g,m,v} 15 lin1.W 16 lin1.b 17 lin2.W 18 lin2.b
//  enc:     19 W_hh 20 W_ih 21 b_hh 22 b_ih
//  pred:    23..26 bn{b,g,m,v} 27 lin1.W 28 lin1.b 29 lin2.W 30 lin2.b
//  rssi2h:  31..42 (same layout as drssi2h)
//  wl:      43..46 bn{b,g,m,v} 47 lin1.W 48 lin1.b 49 lin2.W 50 lin2.b
//  51 k  52 sign
// ---------------------------------------------------------------------------
extern "C" void kernel_launch(void* const* d_in, const int* in_sizes, int n_in,
                              void* d_out, int out_size, void* d_ws, size_t ws_size,
                              hipStream_t stream)
{
    (void)in_sizes; (void)n_in; (void)out_size; (void)ws_size;
    auto F = [&](int i) { return (const float*)d_in[i]; };

    const float* RSSI  = F(0);
    const float* DRSSI = F(1);
    const float* REF   = F(2);
    const float *dWhh = F(3), *dWih = F(4), *dbhh = F(5), *dbih = F(6);
    const float *d2_bn1b = F(7),  *d2_bn1g = F(8),  *d2_bn1m = F(9),  *d2_bn1v = F(10);
    const float *d2_bn2b = F(11), *d2_bn2g = F(12), *d2_bn2m = F(13), *d2_bn2v = F(14);
    const float *d2_W1 = F(15), *d2_b1 = F(16), *d2_W2 = F(17), *d2_b2 = F(18);
    const float *eWhh = F(19), *eWih = F(20), *ebhh = F(21), *ebih = F(22);
    const float *p_bnb = F(23), *p_bng = F(24), *p_bnm = F(25), *p_bnv = F(26);
    const float *p_W1 = F(27), *p_b1 = F(28), *p_W2 = F(29), *p_b2 = F(30);
    const float *r_bn1b = F(31), *r_bn1g = F(32), *r_bn1m = F(33), *r_bn1v = F(34);
    const float *r_bn2b = F(35), *r_bn2g = F(36), *r_bn2m = F(37), *r_bn2v = F(38);
    const float *r_W1 = F(39), *r_b1 = F(40), *r_W2 = F(41), *r_b2 = F(42);
    const float *w_bnb = F(43), *w_bng = F(44), *w_bnm = F(45), *w_bnv = F(46);
    const float *w_W1 = F(47), *w_b1 = F(48), *w_W2 = F(49);

    // ---- workspace: f32 activations, then packed f16 weights
    float* ws = (float*)d_ws;
    size_t o = 0;
    float* enc_in = ws + o; o += (size_t)B_ * T_ * 120;
    float* tr_all = ws + o; o += (size_t)B_ * T_ * 120;
    float* Hbuf   = ws + o; o += (size_t)B_ * T_ * 60;
    float* hb     = ws + o; o += (size_t)B_ * 60;
    float* cb     = ws + o; o += (size_t)B_ * 60;
    float* Cib    = ws + o; o += (size_t)B_ * 60;
    float* yhb    = ws + o; o += (size_t)B_ * 120;
    float* Yb     = ws + o; o += (size_t)B_ * 120;
    _Float16* Zh  = (_Float16*)(ws + o); o += (size_t)B_ * 120;  // B*240 halves
    o = (o + 7) & ~(size_t)7;                        // 16B-align f16 region
    _Float16* wsH = (_Float16*)(ws + o);
    size_t oh = 0;
    _Float16* d2W1p = wsH + oh; oh += 240 * 128;
    _Float16* d2W2p = wsH + oh; oh += 128 * 256;
    _Float16* rW1p  = wsH + oh; oh += 240 * 128;
    _Float16* rW2p  = wsH + oh; oh += 128 * 256;
    _Float16* wlW1p = wsH + oh; oh += 96 * 128;
    _Float16* pW1p  = wsH + oh; oh += 240 * 384;
    _Float16* pW2p  = wsH + oh; oh += 1008 * 256;
    _Float16* eWp   = wsH + oh; oh += 240 * 192;
    _Float16* dWp   = wsH + oh; oh += 240 * 384;

    dim3 blk(256);
    float* outp = (float*)d_out;

    // ---- pre-pack all weights to padded f16 (loop-invariant, ~1.2 MB)
    auto pk = [&](const float* W, int out, int kin, _Float16* Wp, int outP, int KP) {
        int nel = outP * KP;
        pack_w_kernel<<<(nel + 255) / 256, blk, 0, stream>>>(W, out, kin, Wp, outP, KP);
    };
    pk(d2_W1, 240, 120, d2W1p, 240, 128);
    pk(d2_W2, 120, 240, d2W2p, 128, 256);
    pk(r_W1,  240, 120, rW1p,  240, 128);
    pk(r_W2,  120, 240, rW2p,  128, 256);
    pk(w_W1,   90, 120, wlW1p,  96, 128);
    pk(p_W1,  240, 360, pW1p,  240, 384);
    pk(p_W2, 1000, 240, pW2p, 1008, 256);
    pack_w2_kernel<<<(240 * 192 + 255) / 256, blk, 0, stream>>>(eWih, 120, eWhh, 60, eWp, 240, 192);
    pack_w2_kernel<<<(240 * 384 + 255) / 256, blk, 0, stream>>>(dWih, 300, dWhh, 60, dWp, 240, 384);

    // ---- enc_in: head (drssi2h on DRSSI[:, :15]) + tail (rssi2h on DRSSI[:, 15])
    feat_kernel<<<(B_ * 15) / 64, blk, 0, stream>>>(
        DRSSI, 0, 15, 16, B_ * 15,
        d2_bn1b, d2_bn1g, d2_bn1m, d2_bn1v, d2W1p, d2_b1,
        d2_bn2b, d2_bn2g, d2_bn2m, d2_bn2v, d2W2p, d2_b2,
        enc_in, 0, 15, 16);
    feat_kernel<<<B_ / 64, blk, 0, stream>>>(
        DRSSI, 15LL * 120, 1, 16, B_,
        r_bn1b, r_bn1g, r_bn1m, r_bn1v, rW1p, r_b1,
        r_bn2b, r_bn2g, r_bn2m, r_bn2v, rW2p, r_b2,
        enc_in, 15LL * 120, 1, 16);

    // ---- tr_all = feat(RSSI) for all timesteps (loop-invariant precompute)
    feat_kernel<<<(B_ * T_) / 64, blk, 0, stream>>>(
        RSSI, 0, 1, 1, B_ * T_,
        r_bn1b, r_bn1g, r_bn1m, r_bn1v, rW1p, r_b1,
        r_bn2b, r_bn2g, r_bn2m, r_bn2v, rW2p, r_b2,
        tr_all, 0, 1, 1);

    hipMemsetAsync(hb, 0, (size_t)B_ * 60 * sizeof(float), stream);
    hipMemsetAsync(cb, 0, (size_t)B_ * 60 * sizeof(float), stream);
    hipMemsetAsync(Yb, 0, (size_t)B_ * 120 * sizeof(float), stream);

    // ---- encoder: 16 sequential LSTM steps (KP=192)
    for (int t = 0; t < T_; t++) {
        lstm_kernel<192><<<B_ / 32, blk, 0, stream>>>(
            enc_in, (long long)t * 120, T_ * 120, 120,
            nullptr, 0, 0, 0, nullptr, 0, 0, 0,
            hb, cb, eWp, ebih, ebhh, 120, B_,
            Hbuf, (long long)t * 60, T_ * 60);
    }

    // ---- decoder: 16 data-dependent steps (argmax gather feeds next step)
    for (int t = 0; t < T_; t++) {
        attn_kernel<<<B_ / 4, blk, 0, stream>>>(
            Hbuf, hb, w_bnb, w_bng, w_bnm, w_bnv, wlW1p, w_b1, w_W2, Cib);

        feat_kernel<<<B_ / 64, blk, 0, stream>>>(
            Yb, 0, 1, 1, B_,
            r_bn1b, r_bn1g, r_bn1m, r_bn1v, rW1p, r_b1,
            r_bn2b, r_bn2g, r_bn2m, r_bn2v, rW2p, r_b2,
            yhb, 0, 1, 1);

        lstm_kernel<384><<<B_ / 32, blk, 0, stream>>>(
            yhb, 0, 120, 120,
            tr_all, (long long)t * 120, T_ * 120, 120,
            Cib, 0, 60, 60,
            hb, cb, dWp, dbih, dbhh, 300, B_,
            nullptr, 0, 0);

        pred1_kernel<<<B_ / 32, blk, 0, stream>>>(
            yhb, hb, tr_all, (long long)t * 120, T_ * 120, Cib,
            p_bnb, p_bng, p_bnm, p_bnv, pW1p, p_b1, Zh, B_);

        pred2_kernel<<<B_ / 64, blk, 0, stream>>>(
            Zh, pW2p, p_b2, outp, (long long)t * CLS_, T_ * CLS_, B_);

        argmax_gather_kernel<<<B_ / 8, blk, 0, stream>>>(
            outp, (long long)t * CLS_, T_ * CLS_, REF, Yb, B_);
    }
}
